// SimpleAttention_19988777796227
// MI455X (gfx1250) — matrix operand order repaired and checked by reading, not compiled, not run on hardware
//
#include <hip/hip_runtime.h>
#include <hip/hip_bf16.h>

// ---------------------------------------------------------------------------
// SimpleAttention forward for MI455X (gfx1250, wave32, WMMA, TDM staging).
// Pipeline (all on `stream`):
//   0. cvt       : x, Wq, Wk, Wv, Wo f32 -> bf16 (one-time, bandwidth-bound)
//   1. cache_prep: cached_k/v f32 -> d_out k/v sections; bf16 copies (v transposed)
//   2. gemm<1>   : q   = x @ Wq^T -> q_bf16
//   3. gemm<2>   : k_n = x @ Wk^T -> f32 d_out(k, t>=2048) + bf16 kbf
//   4. gemm<3>   : v_n = x @ Wv^T -> f32 d_out(v, t>=2048) + bf16 vT (transposed)
//   5. attn      : flash attention (online softmax) -> O_bf16
//   6. gemm<0>   : out = O @ Wo^T -> f32 d_out
// GEMM: 128x128 block, BK=64, double-buffered LDS. Staging:
//   - if tensor_load_to_lds builtin exists: TDM DMA (wave 0 only, TENSORcnt,
//     hardware LDS padding 128B->144B rows); arity per toolchain
//   - otherwise: per-thread global_load_async_to_lds_b128 (ASYNCcnt)
// 16 v_wmma_f32_16x16x32_bf16 per barrier. Workspace: ~151 MB bf16 scratch.
// ---------------------------------------------------------------------------

typedef __attribute__((ext_vector_type(16))) __bf16 v16bf;
typedef __attribute__((ext_vector_type(8)))  float  v8f;
typedef __attribute__((ext_vector_type(4)))  unsigned int v4u;
typedef __attribute__((ext_vector_type(8)))  int v8i;
typedef __attribute__((ext_vector_type(4)))  int v4i;

// ---- compile-time probes (feedback via stderr) ----
#define TDM_STR2(x) #x
#define TDM_STR(x) TDM_STR2(x)
#pragma message("PROBE clang_major=" TDM_STR(__clang_major__))
#if __has_builtin(__builtin_amdgcn_tensor_load_to_lds)
#pragma message("PROBE has_tensor_load_to_lds=1")
#define USE_TDM 1
#else
#pragma message("PROBE has_tensor_load_to_lds=0 (async fallback)")
#define USE_TDM 0
#endif

union FragAB {            // 32 bytes: one WMMA A or B operand (16 bf16 per lane)
  v16bf v;
  uint4 u[2];
};

__device__ __forceinline__ unsigned short f32_to_bf16(float f) {
  unsigned int u = __float_as_uint(f);
  unsigned int r = u + 0x7FFFu + ((u >> 16) & 1u);   // round-to-nearest-even
  return (unsigned short)(r >> 16);
}
__device__ __forceinline__ unsigned int pack2_bf16(float a, float b) {
  return (unsigned int)f32_to_bf16(a) | ((unsigned int)f32_to_bf16(b) << 16);
}
__device__ __forceinline__ v8f v8zero() {
  v8f z;
#pragma unroll
  for (int i = 0; i < 8; ++i) z[i] = 0.f;
  return z;
}
__device__ __forceinline__ v8f wmma_bf16(v16bf a, v16bf b, v8f c) {
  return __builtin_amdgcn_wmma_f32_16x16x32_bf16(false, a, false, b, (short)0, c,
                                                 false, false);
}
// Low 32 bits of a generic pointer into __shared__ == LDS byte offset
// (flat LDS aperture: the LDS address is the low 32 bits, ISA 10.2).
__device__ __forceinline__ unsigned lds_off32(const void* p) {
  return (unsigned)(unsigned long long)(uintptr_t)p;
}

#if USE_TDM
// TDM D# (ISA 8.3-8.5): load a 128-row x 64-elem bf16 tile (rows of 128B)
// into LDS with +16B padding per row (final LDS row stride 144B = 72 elems).
__device__ __forceinline__ void tdm_load_tile(unsigned lds_addr,
                                              const void* gaddr) {
  unsigned long long ga = (unsigned long long)(uintptr_t)gaddr;
  v4u g0;
  g0[0] = 1u;                                   // count=1 (valid), user mode
  g0[1] = lds_addr;                             // lds_addr [63:32]
  g0[2] = (unsigned)ga;                         // global_addr lo
  g0[3] = (unsigned)(ga >> 32) | 0x80000000u;   // global_addr hi | type=2
  v8i g1;
  g1[0] = 0x07110000;   // data_size=2B | pad_en | pad_interval=4(128B) | pad_amount=3(16B)
  g1[1] = (int)(2048u << 16);   // tensor_dim0 = 2048 (elems per line)
  g1[2] = (int)(128u << 16);    // tensor_dim1 = 128 rows (tile-exact)
  g1[3] = (int)(64u << 16);     // tile_dim0 = 64 elems (128B)
  g1[4] = 128;                  // tile_dim1 = 128 rows, tile_dim2 = 0
  g1[5] = 2048;                 // tensor_dim0_stride = K
  g1[6] = 0;
  g1[7] = 0;
  v4i gz = {0, 0, 0, 0};
#if __clang_major__ >= 23
  v8i gz8 = {0, 0, 0, 0, 0, 0, 0, 0};
  __builtin_amdgcn_tensor_load_to_lds(g0, g1, gz, gz, gz8, 0);
#else
  __builtin_amdgcn_tensor_load_to_lds(g0, g1, gz, gz, 0);
#endif
}
#endif

#define BDIM   2
#define SDIM   2048
#define DDIM   2048
#define HDIM   16
#define DH     128
#define CACHE  2048
#define TDIM   4096
#define MDIM   4096

// ---------------------------------------------------------------------------
// 0) f32 -> bf16 conversion (4 elems/thread)
// ---------------------------------------------------------------------------
__global__ __launch_bounds__(256) void cvt_f32_bf16_kernel(
    const float* __restrict__ in, unsigned short* __restrict__ out, int n4) {
  int i = blockIdx.x * 256 + threadIdx.x;
  if (i < n4) {
    float4 f = ((const float4*)in)[i];
    uint2 p;
    p.x = pack2_bf16(f.x, f.y);
    p.y = pack2_bf16(f.z, f.w);
    ((uint2*)out)[i] = p;
  }
}

// ---------------------------------------------------------------------------
// 1) Cache prep
// ---------------------------------------------------------------------------
__global__ __launch_bounds__(256) void cache_prep_kernel(
    const float* __restrict__ ck, const float* __restrict__ cv,
    float* __restrict__ outk, float* __restrict__ outv,
    unsigned short* __restrict__ kbf, unsigned short* __restrict__ vT) {
  int idx = blockIdx.x * 256 + threadIdx.x;          // < B*H*CACHE*DH
  int dh = idx & 127;
  int t  = (idx >> 7) & (CACHE - 1);
  int bh = idx >> 18;
  float kvl = ck[idx];
  float vvl = cv[idx];
  size_t oi = ((size_t)bh * TDIM + t) * DH + dh;
  outk[oi] = kvl;
  kbf[oi]  = f32_to_bf16(kvl);
  outv[oi] = vvl;
  vT[((size_t)bh * DH + dh) * TDIM + t] = f32_to_bf16(vvl);
}

// ---------------------------------------------------------------------------
// 2) GEMM: C[m,n] = sum_k A[m,k]*B[n,k], all-bf16 operands, M=4096 N=K=2048.
//    Double-buffered LDS; TDM (or async-VMEM) staging; BK=64.
// ---------------------------------------------------------------------------
template <int MODE>
__global__ __launch_bounds__(256) void gemm_bf16_wmma(
    const unsigned short* __restrict__ Abf, const unsigned short* __restrict__ Bbf,
    float* __restrict__ outF, unsigned short* __restrict__ outH) {
  constexpr int Kdim = DDIM, Ndim = DDIM;
  constexpr int BK = 64;
  constexpr int LDP = 72;                   // padded row stride (bf16 elems)
  __shared__ unsigned short As[2][128 * LDP];
  __shared__ unsigned short Bs[2][128 * LDP];

  const int tid  = threadIdx.x;
  const int lane = tid & 31;
  const int wave = tid >> 5;
  const int wm   = wave >> 1;               // 0..3 (M)
  const int wn   = wave & 1;                // 0..1 (N)
  const int bM   = blockIdx.y * 128;
  const int bN   = blockIdx.x * 128;
  const int ar   = lane & 15;
  const int ah   = lane >> 4;

#if USE_TDM
  const unsigned ldsA[2] = {lds_off32(&As[0][0]), lds_off32(&As[1][0])};
  const unsigned ldsB[2] = {lds_off32(&Bs[0][0]), lds_off32(&Bs[1][0])};
  const unsigned short* agt = Abf + (size_t)bM * Kdim;
  const unsigned short* bgt = Bbf + (size_t)bN * Kdim;
#else
  const int srow = tid >> 1;                // staging row 0..127
  const int scol = (tid & 1) * 32;          // staging col block (32 elems = 64B)
  const unsigned short* ag0 = Abf + (size_t)(bM + srow) * Kdim + scol;
  const unsigned short* bg0 = Bbf + (size_t)(bN + srow) * Kdim + scol;
  const unsigned laddrA[2] = {lds_off32(&As[0][srow * LDP + scol]),
                              lds_off32(&As[1][srow * LDP + scol])};
  const unsigned laddrB[2] = {lds_off32(&Bs[0][srow * LDP + scol]),
                              lds_off32(&Bs[1][srow * LDP + scol])};
#endif

  auto stage = [&](int p, int kk) {
#if USE_TDM
    // One DMA per matrix tile; EXEC-independent, issued by wave 0 only.
    tdm_load_tile(ldsA[p], agt + kk);
    tdm_load_tile(ldsB[p], bgt + kk);
#else
    unsigned long long ga = (unsigned long long)(uintptr_t)(ag0 + kk);
    unsigned long long gb = (unsigned long long)(uintptr_t)(bg0 + kk);
    asm volatile("global_load_async_to_lds_b128 %0, %1, off"
                 :: "v"(laddrA[p]), "v"(ga) : "memory");
    asm volatile("global_load_async_to_lds_b128 %0, %1, off offset:32"
                 :: "v"(laddrA[p]), "v"(ga) : "memory");
    asm volatile("global_load_async_to_lds_b128 %0, %1, off"
                 :: "v"(laddrB[p]), "v"(gb) : "memory");
    asm volatile("global_load_async_to_lds_b128 %0, %1, off offset:32"
                 :: "v"(laddrB[p]), "v"(gb) : "memory");
#endif
  };

  v8f acc[2][4];
#pragma unroll
  for (int i = 0; i < 2; ++i)
#pragma unroll
    for (int j = 0; j < 4; ++j) acc[i][j] = v8zero();

#if USE_TDM
  if (wave == 0) stage(0, 0);
#else
  stage(0, 0);
#endif

  for (int kk = 0; kk < Kdim; kk += BK) {
    const int p = (kk >> 6) & 1;
#if USE_TDM
    if (wave == 0) __builtin_amdgcn_s_wait_tensorcnt(0);  // DMA landed
#else
    asm volatile("s_wait_asynccnt 0" ::: "memory");
#endif
    __syncthreads();                                      // tile published
    if (kk + BK < Kdim) {
#if USE_TDM
      if (wave == 0) stage(p ^ 1, kk + BK);               // overlap next DMA
#else
      stage(p ^ 1, kk + BK);
#endif
    }

#pragma unroll
    for (int ks = 0; ks < 2; ++ks) {                      // two k=32 steps
      FragAB fa[2], fb[4];
#pragma unroll
      for (int i = 0; i < 2; ++i) {
        const unsigned short* q = &As[p][(wm * 32 + i * 16 + ar) * LDP + ks * 32];
        fa[i].u[0] = *(const uint4*)(q + ah * 16);
        fa[i].u[1] = *(const uint4*)(q + ah * 16 + 8);
      }
#pragma unroll
      for (int j = 0; j < 4; ++j) {
        const unsigned short* q =
            &Bs[p][(wn * 64 + j * 16 + ar) * LDP + ks * 32 + ah * 16];
        fb[j].u[0] = *(const uint4*)q;
        fb[j].u[1] = *(const uint4*)(q + 8);
      }
#pragma unroll
      for (int i = 0; i < 2; ++i)
#pragma unroll
        for (int j = 0; j < 4; ++j)
          acc[i][j] = wmma_bf16(fa[i].v, fb[j].v, acc[i][j]);
    }
  }

  // ---- epilogue ----
#pragma unroll
  for (int i = 0; i < 2; ++i) {
#pragma unroll
    for (int j = 0; j < 4; ++j) {
      const int mbase = bM + wm * 32 + i * 16 + ah * 8;
      const int n     = bN + wn * 64 + j * 16 + ar;
#pragma unroll
      for (int r = 0; r < 8; ++r) {
        const int m = mbase + r;
        const float val = acc[i][j][r];
        if constexpr (MODE == 0) {
          outF[(size_t)m * Ndim + n] = val;
        } else if constexpr (MODE == 1) {
          outH[(size_t)m * Ndim + n] = f32_to_bf16(val);
        } else {
          const int b  = m >> 11;
          const int s  = m & (SDIM - 1);
          const int hh = n >> 7;
          const int dh = n & (DH - 1);
          const size_t ik =
              (((size_t)(b * HDIM + hh)) * TDIM + CACHE + s) * DH + dh;
          outF[ik] = val;
          if constexpr (MODE == 2) {
            outH[ik] = f32_to_bf16(val);
          } else {
            outH[(((size_t)(b * HDIM + hh)) * DH + dh) * TDIM + (CACHE + s)] =
                f32_to_bf16(val);
          }
        }
      }
    }
  }
}

// ---------------------------------------------------------------------------
// 5) Flash attention: each wave owns a 16-row Q block of one (b,h);
//    32-key chunks: 8 score WMMAs + online softmax + 8 PV WMMAs.
// ---------------------------------------------------------------------------
__global__ __launch_bounds__(256) void attn_fwd_kernel(
    const unsigned short* __restrict__ qbf, const unsigned short* __restrict__ kbf,
    const unsigned short* __restrict__ vT, unsigned short* __restrict__ obf) {
  __shared__ unsigned short Plds[8 * 16 * 40];

  const int lane = threadIdx.x & 31;
  const int wave = threadIdx.x >> 5;
  const int ar   = lane & 15;
  const int ah   = lane >> 4;
  const int b  = blockIdx.z;
  const int h  = blockIdx.y;
  const int s0 = blockIdx.x * 128 + wave * 16;
  const int bh = b * HDIM + h;

  const unsigned short* qbase = qbf + (size_t)(b * SDIM + s0) * DDIM + h * DH;
  FragAB qf[4];
#pragma unroll
  for (int kc = 0; kc < 4; ++kc) {
    const unsigned short* qp = qbase + (size_t)ar * DDIM + kc * 32;
    qf[kc].u[0] = *(const uint4*)(qp + ah * 16);
    qf[kc].u[1] = *(const uint4*)(qp + ah * 16 + 8);
  }

  float mr[8], lr[8];
  v8f oacc[8];
#pragma unroll
  for (int r = 0; r < 8; ++r) { mr[r] = -3.0e38f; lr[r] = 0.f; }
#pragma unroll
  for (int j = 0; j < 8; ++j) oacc[j] = v8zero();

  const unsigned short* kb0 = kbf + (size_t)bh * TDIM * DH;
  const unsigned short* vb0 = vT + (size_t)bh * DH * TDIM;
  unsigned short* pw = &Plds[wave * 16 * 40];
  const float scale = 0.08838834764831845f;   // 1/sqrt(128)

  for (int t0 = 0; t0 < TDIM; t0 += 32) {
    v8f sc0 = v8zero(), sc1 = v8zero();
#pragma unroll
    for (int kc = 0; kc < 4; ++kc) {
      const unsigned short* kp0 = kb0 + (size_t)(t0 + ar) * DH + kc * 32 + ah * 16;
      FragAB f0;
      f0.u[0] = *(const uint4*)kp0;
      f0.u[1] = *(const uint4*)(kp0 + 8);
      sc0 = wmma_bf16(qf[kc].v, f0.v, sc0);
      const unsigned short* kp1 =
          kb0 + (size_t)(t0 + 16 + ar) * DH + kc * 32 + ah * 16;
      FragAB f1;
      f1.u[0] = *(const uint4*)kp1;
      f1.u[1] = *(const uint4*)(kp1 + 8);
      sc1 = wmma_bf16(qf[kc].v, f1.v, sc1);
    }

    float corr[8];
#pragma unroll
    for (int r = 0; r < 8; ++r) {
      float a = sc0[r] * scale;
      float c = sc1[r] * scale;
      float pm = fmaxf(a, c);
#pragma unroll
      for (int off = 1; off < 16; off <<= 1) pm = fmaxf(pm, __shfl_xor(pm, off));
      const float mn = fmaxf(mr[r], pm);
      corr[r] = __expf(mr[r] - mn);
      mr[r] = mn;
      const float e0 = __expf(a - mn);
      const float e1 = __expf(c - mn);
      sc0[r] = e0;
      sc1[r] = e1;
      float rs = e0 + e1;
#pragma unroll
      for (int off = 1; off < 16; off <<= 1) rs += __shfl_xor(rs, off);
      lr[r] = lr[r] * corr[r] + rs;
    }
#pragma unroll
    for (int j = 0; j < 8; ++j)
#pragma unroll
      for (int r = 0; r < 8; ++r) oacc[j][r] *= corr[r];

    // P: C-layout -> A-layout via per-wave LDS (intra-wave, DScnt-guarded)
#pragma unroll
    for (int r = 0; r < 8; ++r) {
      pw[(r + 8 * ah) * 40 + ar]      = f32_to_bf16(sc0[r]);
      pw[(r + 8 * ah) * 40 + 16 + ar] = f32_to_bf16(sc1[r]);
    }
    asm volatile("s_wait_dscnt 0" ::: "memory");
    FragAB pf;
    {
      const unsigned short* pr = pw + ar * 40;
      pf.u[0] = *(const uint4*)(pr + ah * 16);
      pf.u[1] = *(const uint4*)(pr + ah * 16 + 8);
    }

#pragma unroll
    for (int j = 0; j < 8; ++j) {
      const unsigned short* vp =
          vb0 + (size_t)(j * 16 + ar) * TDIM + t0 + ah * 16;
      FragAB fv;
      fv.u[0] = *(const uint4*)vp;
      fv.u[1] = *(const uint4*)(vp + 8);
      oacc[j] = wmma_bf16(pf.v, fv.v, oacc[j]);
    }
  }

  unsigned short* ob = obf + (size_t)(b * SDIM + s0) * DDIM + h * DH;
#pragma unroll
  for (int r = 0; r < 8; ++r) {
    const float inv = 1.f / lr[r];
#pragma unroll
    for (int j = 0; j < 8; ++j) {
      ob[(size_t)(r + 8 * ah) * DDIM + j * 16 + ar] =
          f32_to_bf16(oacc[j][r] * inv);
    }
  }
}

// ---------------------------------------------------------------------------
// Launch
// ---------------------------------------------------------------------------
extern "C" void kernel_launch(void* const* d_in, const int* in_sizes, int n_in,
                              void* d_out, int out_size, void* d_ws, size_t ws_size,
                              hipStream_t stream) {
  const float* x  = (const float*)d_in[0];
  const float* ck = (const float*)d_in[1];
  const float* cv = (const float*)d_in[2];
  const float* Wq = (const float*)d_in[3];
  const float* Wk = (const float*)d_in[4];
  const float* Wv = (const float*)d_in[5];
  const float* Wo = (const float*)d_in[6];

  float* out  = (float*)d_out;
  float* outk = out + (size_t)BDIM * SDIM * DDIM;
  float* outv = outk + (size_t)BDIM * HDIM * TDIM * DH;

  const size_t nX = (size_t)BDIM * SDIM * DDIM;      // 8.4M
  const size_t nW = (size_t)DDIM * DDIM;             // 4.2M
  const size_t nKV = (size_t)BDIM * HDIM * TDIM * DH;// 16.8M

  unsigned short* xbf  = (unsigned short*)d_ws;
  unsigned short* wqbf = xbf  + nX;
  unsigned short* wkbf = wqbf + nW;
  unsigned short* wvbf = wkbf + nW;
  unsigned short* wobf = wvbf + nW;
  unsigned short* qbf  = wobf + nW;
  unsigned short* kbf  = qbf  + nX;
  unsigned short* vTb  = kbf  + nKV;
  unsigned short* obf  = vTb  + nKV;

  cvt_f32_bf16_kernel<<<(int)(nX / 4 / 256), 256, 0, stream>>>(x, xbf, (int)(nX / 4));
  cvt_f32_bf16_kernel<<<(int)(nW / 4 / 256), 256, 0, stream>>>(Wq, wqbf, (int)(nW / 4));
  cvt_f32_bf16_kernel<<<(int)(nW / 4 / 256), 256, 0, stream>>>(Wk, wkbf, (int)(nW / 4));
  cvt_f32_bf16_kernel<<<(int)(nW / 4 / 256), 256, 0, stream>>>(Wv, wvbf, (int)(nW / 4));
  cvt_f32_bf16_kernel<<<(int)(nW / 4 / 256), 256, 0, stream>>>(Wo, wobf, (int)(nW / 4));
  cache_prep_kernel<<<32768, 256, 0, stream>>>(ck, cv, outk, outv, kbf, vTb);

  gemm_bf16_wmma<1><<<dim3(16, 32), 256, 0, stream>>>(xbf, wqbf, nullptr, qbf);
  gemm_bf16_wmma<2><<<dim3(16, 32), 256, 0, stream>>>(xbf, wkbf, outk, kbf);
  gemm_bf16_wmma<3><<<dim3(16, 32), 256, 0, stream>>>(xbf, wvbf, outv, vTb);
  attn_fwd_kernel<<<dim3(16, 16, 2), 256, 0, stream>>>(qbf, kbf, vTb, obf);
  gemm_bf16_wmma<0><<<dim3(16, 32), 256, 0, stream>>>(obf, wobf, out, nullptr);
}
